// GaussianHeadModule_41549513621844
// MI455X (gfx1250) — matrix operand order repaired and checked by reading, not compiled, not run on hardware
//
#include <hip/hip_runtime.h>
#include <hip/hip_bf16.h>

// ---------------------------------------------------------------------------
// GaussianHead forward for gfx1250 (MI455X): all MLP math via
// v_wmma_f32_16x16x32_f16 (wave32, 16x16x32 tiles), activations staged in LDS,
// per-batch-constant input slices folded into layer-0 biases.
// ---------------------------------------------------------------------------

#define LRELU_K 0.2f
#define NEAR_D 0.1f
#define FAR_D 0.25f
#define DEFORM_SCALE 0.3f
#define N_LMK 66

typedef __attribute__((ext_vector_type(8)))  _Float16 v8h;
typedef __attribute__((ext_vector_type(16))) _Float16 v16h;
typedef __attribute__((ext_vector_type(8)))  float    v8f;

__device__ __forceinline__ v16h cat8(v8h a, v8h b) {
  return __builtin_shufflevector(a, b, 0,1,2,3,4,5,6,7,8,9,10,11,12,13,14,15);
}

// One 16x16 f32 tile: A = 16 points x K (LDS, row-major, lda halves/row),
// B = W rows (16 output features) x K (global f16, row-major, ldw halves/row).
// A fragment layout (16-bit A 16x32): lanes 0-15 K=[k0+0..7],[k0+16..23];
// lanes 16-31 K=[k0+8..15],[k0+24..31].  B fragment: same K order as A
// (lanes 0-15 K=[k0+0..7],[k0+16..23]; lanes 16-31 K=[k0+8..15],[k0+24..31]).
__device__ __forceinline__ v8f mma_tile(const _Float16* __restrict__ W, int ldw,
                                        const _Float16* A, int lda,
                                        int kdim, int lane)
{
  const int n15 = lane & 15;
  const int sel = lane >> 4;
  v8f c = {0.f, 0.f, 0.f, 0.f, 0.f, 0.f, 0.f, 0.f};
  for (int k0 = 0; k0 < kdim; k0 += 32) {
    const _Float16* ap = A + n15 * lda + k0 + sel * 8;
    v8h a0 = *(const v8h*)(ap);
    v8h a1 = *(const v8h*)(ap + 16);
    const _Float16* bp = W + n15 * ldw + k0 + sel * 8;
    v8h b0 = *(const v8h*)(bp);
    v8h b1 = *(const v8h*)(bp + 16);
    c = __builtin_amdgcn_wmma_f32_16x16x32_f16(false, cat8(a0, a1),
                                               false, cat8(b0, b1),
                                               (short)0, c, false, false);
  }
  return c;
}

// C/D layout: VGPR r -> row M = r + 8*(lane>=16), col N = lane&15.
__device__ __forceinline__ void store_act(_Float16* dst, int ldd,
                                          const float* __restrict__ bias,
                                          v8f c, int lane)
{
  const int n = lane & 15, sel = lane >> 4;
  const float bv = bias[n];
#pragma unroll
  for (int r = 0; r < 8; ++r) {
    float v = c[r] + bv;
    v = v > 0.f ? v : LRELU_K * v;
    dst[(r + 8 * sel) * ldd + n] = (_Float16)v;
  }
}

// -------------------------- weight conversion ------------------------------
__global__ void cvt_w(const float* __restrict__ src, _Float16* __restrict__ dst,
                      int srcRows, int srcCols, int srcLd, int dstRows, int dstCols)
{
  int i = blockIdx.x * blockDim.x + threadIdx.x;
  if (i >= dstRows * dstCols) return;
  int r = i / dstCols, c = i - r * dstCols;
  float v = (r < srcRows && c < srcCols) ? src[r * srcLd + c] : 0.f;
  dst[i] = (_Float16)v;
}

// -------------------------- per-batch prep ---------------------------------
__global__ void prep_batch(const float* __restrict__ pose, const float* __restrict__ exp_coeff,
                           const float* __restrict__ scale,
                           const float* __restrict__ Wec0, const float* __restrict__ bec0r,
                           const float* __restrict__ Wpc0, const float* __restrict__ bpc0r,
                           const float* __restrict__ Wed0, const float* __restrict__ bed0r,
                           const float* __restrict__ Wpd0, const float* __restrict__ bpd0r,
                           const float* __restrict__ bed2r, const float* __restrict__ bpd1r,
                           float* __restrict__ Rw, float* __restrict__ Tw, float* __restrict__ Sw,
                           float* __restrict__ bec0, float* __restrict__ bpc0,
                           float* __restrict__ bed0, float* __restrict__ bpd0,
                           float* __restrict__ bed2p, float* __restrict__ bpd1p)
{
  const int b = blockIdx.x;
  const int t = threadIdx.x;
  __shared__ float pe[54];
  __shared__ float ex[64];
  if (t < 54) {
    if (t < 6) pe[t] = pose[b * 6 + t];
    else {
      int k = t - 6, fi = k / 12, r2 = k % 12;
      float v = pose[b * 6 + (r2 % 6)] * (float)(1 << fi);
      pe[t] = (r2 < 6) ? sinf(v) : cosf(v);
    }
  }
  if (t < 64) ex[t] = exp_coeff[b * 64 + t];
  __syncthreads();
  if (t == 0) {
    float w0 = pose[b * 6], w1 = pose[b * 6 + 1], w2 = pose[b * 6 + 2];
    float th = sqrtf(w0 * w0 + w1 * w1 + w2 * w2);
    if (th < 1e-6f) th = 1e-6f;
    float Af = sinf(th) / th, Bf = (1.f - cosf(th)) / (th * th);
    float K[3][3] = {{0.f, -w2, w1}, {w2, 0.f, -w0}, {-w1, w0, 0.f}};
    float K2[3][3];
#pragma unroll
    for (int i2 = 0; i2 < 3; ++i2)
#pragma unroll
      for (int j2 = 0; j2 < 3; ++j2)
        K2[i2][j2] = K[i2][0] * K[0][j2] + K[i2][1] * K[1][j2] + K[i2][2] * K[2][j2];
#pragma unroll
    for (int i2 = 0; i2 < 3; ++i2)
#pragma unroll
      for (int j2 = 0; j2 < 3; ++j2)
        Rw[b * 9 + i2 * 3 + j2] = (i2 == j2 ? 1.f : 0.f) + Af * K[i2][j2] + Bf * K2[i2][j2];
    Tw[b * 3 + 0] = pose[b * 6 + 3];
    Tw[b * 3 + 1] = pose[b * 6 + 4];
    Tw[b * 3 + 2] = pose[b * 6 + 5];
    Sw[b] = scale[b];
    if (b == 0) {
      for (int c = 0; c < 16; ++c) {
        bed2p[c] = (c < 3) ? bed2r[c] : 0.f;
        bpd1p[c] = (c < 3) ? bpd1r[c] : 0.f;
      }
    }
  }
  // fold per-batch-constant input columns into layer-0 biases
  for (int o = t; o < 256; o += blockDim.x) {
    float s = bec0r[o];
    for (int j = 0; j < 64; ++j) s += Wec0[o * 192 + 128 + j] * ex[j];
    bec0[b * 256 + o] = s;
    float s2 = bed0r[o];
    for (int j = 0; j < 64; ++j) s2 += Wed0[o * 91 + 27 + j] * ex[j];
    bed0[b * 256 + o] = s2;
  }
  for (int o = t; o < 128; o += blockDim.x) {
    float s = bpc0r[o];
    for (int j = 0; j < 54; ++j) s += Wpc0[o * 182 + 128 + j] * pe[j];
    bpc0[b * 128 + o] = s;
    float s2 = bpd0r[o];
    for (int j = 0; j < 54; ++j) s2 += Wpd0[o * 81 + 27 + j] * pe[j];
    bpd0[b * 128 + o] = s2;
  }
}

// -------------------------- fused main kernel ------------------------------
struct MainP {
  const float *xyz, *feat, *lmk;
  const _Float16 *Wec0, *Wec1, *Wec2, *Wpc0, *Wpc1, *Wed0, *Wed1, *Wed2, *Wpd0, *Wpd1;
  const float *bec0, *bpc0, *bed0, *bpd0;            // folded, per batch
  const float *bec1, *bec2, *bpc1, *bed1, *bed2p, *bpd1p;
  const float *R, *T, *S;
  float* out;
  int N;
};

__global__ __launch_bounds__(512) void gh_main(MainP P)
{
  __shared__ _Float16 ft[32][128];   // tanh(feature)
  __shared__ _Float16 xe[32][32];    // posenc(xyz), padded 27->32
  __shared__ _Float16 h1[32][256];
  __shared__ _Float16 h2[32][256];
  __shared__ _Float16 hp[32][128];
  __shared__ _Float16 hpd[32][128];
  __shared__ float colorbuf[32][32];
  __shared__ float defE[32][16];
  __shared__ float defP[32][16];
  __shared__ float weL[32], wpL[32];
  __shared__ float xyzl[32][3];

  const int t = threadIdx.x;
  const int wave = t >> 5;
  const int lane = t & 31;
  const int bb = blockIdx.y;
  const int base = blockIdx.x * 32;
  const int N = P.N;
  const int Bt = gridDim.y;

  // ---- P1: stage per-point inputs ----
  for (int i = t; i < 32 * 128; i += 512) {
    int p = i >> 7, cc = i & 127;
    int gp = base + p;
    float v = (gp < N) ? tanhf(P.feat[(size_t)gp * 128 + cc]) : 0.f;
    ft[p][cc] = (_Float16)v;
  }
  if (t < 32) {
    int gp = base + t;
    int gpc = (gp < N) ? gp : (N - 1);
    float x = P.xyz[gpc * 3 + 0], y = P.xyz[gpc * 3 + 1], z = P.xyz[gpc * 3 + 2];
    xyzl[t][0] = x; xyzl[t][1] = y; xyzl[t][2] = z;
    xe[t][0] = (_Float16)x; xe[t][1] = (_Float16)y; xe[t][2] = (_Float16)z;
#pragma unroll
    for (int fi = 0; fi < 4; ++fi) {
      float f = (float)(1 << fi);
      xe[t][3 + 6 * fi + 0] = (_Float16)sinf(x * f);
      xe[t][3 + 6 * fi + 1] = (_Float16)sinf(y * f);
      xe[t][3 + 6 * fi + 2] = (_Float16)sinf(z * f);
      xe[t][3 + 6 * fi + 3] = (_Float16)cosf(x * f);
      xe[t][3 + 6 * fi + 4] = (_Float16)cosf(y * f);
      xe[t][3 + 6 * fi + 5] = (_Float16)cosf(z * f);
    }
    for (int cth = 27; cth < 32; ++cth) xe[t][cth] = (_Float16)0.f;
    float best = 3.4e38f;
    for (int l = 0; l < N_LMK; ++l) {
      float dx = x - P.lmk[l * 3], dy = y - P.lmk[l * 3 + 1], dz = z - P.lmk[l * 3 + 2];
      float d = dx * dx + dy * dy + dz * dz;
      best = d < best ? d : best;
    }
    float d = best > 0.f ? best : 0.f;
    float ew = (FAR_D - d) / (FAR_D - NEAR_D);
    ew = ew < 0.f ? 0.f : (ew > 1.f ? 1.f : ew);
    weL[t] = (d < FAR_D) ? ew : 0.f;
    wpL[t] = (d > NEAR_D) ? (1.f - ew) : 0.f;
  }
  __syncthreads();

  // ---- P2: exp_color L0 (K=128): ft -> h1 ----
  {
    const int ct = wave;
    const float* bias = P.bec0 + bb * 256 + ct * 16;
    for (int mt = 0; mt < 2; ++mt) {
      v8f c = mma_tile(P.Wec0 + ct * 16 * 128, 128, &ft[mt * 16][0], 128, 128, lane);
      store_act(&h1[mt * 16][ct * 16], 256, bias, c, lane);
    }
  }
  __syncthreads();

  // ---- P3: exp_color L1 (K=256): h1 -> h2 ----
  {
    const int ct = wave;
    const float* bias = P.bec1 + ct * 16;
    for (int mt = 0; mt < 2; ++mt) {
      v8f c = mma_tile(P.Wec1 + ct * 16 * 256, 256, &h1[mt * 16][0], 256, 256, lane);
      store_act(&h2[mt * 16][ct * 16], 256, bias, c, lane);
    }
  }
  __syncthreads();

  // ---- P4 job list: ecL2 (2) | pcL0 (8) | edL0 (16) | pdL0 (8) ----
  for (int j = wave; j < 34; j += 16) {
    if (j < 2) {
      const int ct = j;
      const int n = lane & 15, sel = lane >> 4;
      const float bv = P.bec2[ct * 16 + n];
      for (int mt = 0; mt < 2; ++mt) {
        v8f c = mma_tile(P.Wec2 + ct * 16 * 256, 256, &h2[mt * 16][0], 256, 256, lane);
#pragma unroll
        for (int r = 0; r < 8; ++r)
          colorbuf[mt * 16 + r + 8 * sel][ct * 16 + n] = c[r] + bv;
      }
    } else if (j < 10) {
      const int ct = j - 2;
      const float* bias = P.bpc0 + bb * 128 + ct * 16;
      for (int mt = 0; mt < 2; ++mt) {
        v8f c = mma_tile(P.Wpc0 + ct * 16 * 128, 128, &ft[mt * 16][0], 128, 128, lane);
        store_act(&hp[mt * 16][ct * 16], 128, bias, c, lane);
      }
    } else if (j < 26) {
      const int ct = j - 10;
      const float* bias = P.bed0 + bb * 256 + ct * 16;
      for (int mt = 0; mt < 2; ++mt) {
        v8f c = mma_tile(P.Wed0 + ct * 16 * 32, 32, &xe[mt * 16][0], 32, 32, lane);
        store_act(&h1[mt * 16][ct * 16], 256, bias, c, lane);
      }
    } else {
      const int ct = j - 26;
      const float* bias = P.bpd0 + bb * 128 + ct * 16;
      for (int mt = 0; mt < 2; ++mt) {
        v8f c = mma_tile(P.Wpd0 + ct * 16 * 32, 32, &xe[mt * 16][0], 32, 32, lane);
        store_act(&hpd[mt * 16][ct * 16], 128, bias, c, lane);
      }
    }
  }
  __syncthreads();

  // ---- P5 job list: pcL1 (2, blend+store color) | edL1 (16) ----
  const size_t colorOff = (size_t)Bt * N * 3;
  for (int j = wave; j < 18; j += 16) {
    if (j < 2) {
      const int ct = j;
      const int n = lane & 15, sel = lane >> 4;
      const float bv = P.bpc1[ct * 16 + n];
      for (int mt = 0; mt < 2; ++mt) {
        v8f c = mma_tile(P.Wpc1 + ct * 16 * 128, 128, &hp[mt * 16][0], 128, 128, lane);
#pragma unroll
        for (int r = 0; r < 8; ++r) {
          int p = mt * 16 + r + 8 * sel;
          int gp = base + p;
          if (gp < N) {
            float pc = c[r] + bv;
            float ov = colorbuf[p][ct * 16 + n] * weL[p] + pc * wpL[p];
            P.out[colorOff + ((size_t)bb * N + gp) * 32 + ct * 16 + n] = ov;
          }
        }
      }
    } else {
      const int ct = j - 2;
      const float* bias = P.bed1 + ct * 16;
      for (int mt = 0; mt < 2; ++mt) {
        v8f c = mma_tile(P.Wed1 + ct * 16 * 256, 256, &h1[mt * 16][0], 256, 256, lane);
        store_act(&h2[mt * 16][ct * 16], 256, bias, c, lane);
      }
    }
  }
  __syncthreads();

  // ---- P6: exp_deform L2 (wave 0) / pose_deform L1 (wave 1), tanh -> LDS ----
  if (wave == 0) {
    const int n = lane & 15, sel = lane >> 4;
    const float bv = P.bed2p[n];
    for (int mt = 0; mt < 2; ++mt) {
      v8f c = mma_tile(P.Wed2, 256, &h2[mt * 16][0], 256, 256, lane);
#pragma unroll
      for (int r = 0; r < 8; ++r)
        defE[mt * 16 + r + 8 * sel][n] = tanhf(c[r] + bv);
    }
  } else if (wave == 1) {
    const int n = lane & 15, sel = lane >> 4;
    const float bv = P.bpd1p[n];
    for (int mt = 0; mt < 2; ++mt) {
      v8f c = mma_tile(P.Wpd1, 128, &hpd[mt * 16][0], 128, 128, lane);
#pragma unroll
      for (int r = 0; r < 8; ++r)
        defP[mt * 16 + r + 8 * sel][n] = tanhf(c[r] + bv);
    }
  }
  __syncthreads();

  // ---- P7: blend deform, rigid transform, store xyz ----
  if (t < 32) {
    int gp = base + t;
    if (gp < N) {
      const float Sv = P.S[bb];
      const float* Rb = P.R + bb * 9;
      const float* Tb = P.T + bb * 3;
      float v[3];
#pragma unroll
      for (int jx = 0; jx < 3; ++jx) {
        float dd = defE[t][jx] * weL[t] + defP[t][jx] * wpL[t];
        v[jx] = (xyzl[t][jx] + dd * DEFORM_SCALE) * Sv;
      }
#pragma unroll
      for (int ix = 0; ix < 3; ++ix) {
        float o = Rb[ix * 3 + 0] * v[0] + Rb[ix * 3 + 1] * v[1] + Rb[ix * 3 + 2] * v[2] + Tb[ix];
        P.out[((size_t)bb * N + gp) * 3 + ix] = o;
      }
    }
  }
}

// -------------------------- pointwise outputs ------------------------------
__global__ void gh_pointwise(const float* __restrict__ scales_p, const float* __restrict__ rotation_p,
                             const float* __restrict__ opacity_p,
                             const float* __restrict__ R, const float* __restrict__ S,
                             float* __restrict__ out, int N, int B)
{
  int i = blockIdx.x * blockDim.x + threadIdx.x;
  if (i >= B * N) return;
  int b = i / N, n = i - b * N;
  const size_t scOff = (size_t)B * N * 35;           // after xyz(3)+color(32)
  const size_t rotOff = scOff + (size_t)B * N * 3;
  const size_t opOff = rotOff + (size_t)B * N * 4;
  const float Sv = S[b];
#pragma unroll
  for (int j = 0; j < 3; ++j)
    out[scOff + (size_t)i * 3 + j] = expf(scales_p[n * 3 + j]) * Sv;
  out[opOff + i] = 1.f / (1.f + expf(-opacity_p[n]));

  float q0 = rotation_p[n * 4], q1 = rotation_p[n * 4 + 1];
  float q2 = rotation_p[n * 4 + 2], q3 = rotation_p[n * 4 + 3];
  float nr = sqrtf(q0 * q0 + q1 * q1 + q2 * q2 + q3 * q3);
  nr = nr > 1e-12f ? nr : 1e-12f;
  q0 /= nr; q1 /= nr; q2 /= nr; q3 /= nr;
  float two_s = 2.f / (q0 * q0 + q1 * q1 + q2 * q2 + q3 * q3);
  float r_ = q0, i_ = q1, j_ = q2, k_ = q3;
  float M[3][3];
  M[0][0] = 1.f - two_s * (j_ * j_ + k_ * k_);
  M[0][1] = two_s * (i_ * j_ - k_ * r_);
  M[0][2] = two_s * (i_ * k_ + j_ * r_);
  M[1][0] = two_s * (i_ * j_ + k_ * r_);
  M[1][1] = 1.f - two_s * (i_ * i_ + k_ * k_);
  M[1][2] = two_s * (j_ * k_ - i_ * r_);
  M[2][0] = two_s * (i_ * k_ - j_ * r_);
  M[2][1] = two_s * (j_ * k_ + i_ * r_);
  M[2][2] = 1.f - two_s * (i_ * i_ + j_ * j_);
  const float* Rb = R + b * 9;
  float A[3][3];
#pragma unroll
  for (int ii = 0; ii < 3; ++ii)
#pragma unroll
    for (int kk = 0; kk < 3; ++kk)
      A[ii][kk] = Rb[ii * 3 + 0] * M[0][kk] + Rb[ii * 3 + 1] * M[1][kk] + Rb[ii * 3 + 2] * M[2][kk];
  float m00 = A[0][0], m01 = A[0][1], m02 = A[0][2];
  float m10 = A[1][0], m11 = A[1][1], m12 = A[1][2];
  float m20 = A[2][0], m21 = A[2][1], m22 = A[2][2];
  float qa[4];
  qa[0] = sqrtf(fmaxf(1.f + m00 + m11 + m22, 0.f));
  qa[1] = sqrtf(fmaxf(1.f + m00 - m11 - m22, 0.f));
  qa[2] = sqrtf(fmaxf(1.f - m00 + m11 - m22, 0.f));
  qa[3] = sqrtf(fmaxf(1.f - m00 - m11 + m22, 0.f));
  int k = 0;
  if (qa[1] > qa[k]) k = 1;
  if (qa[2] > qa[k]) k = 2;
  if (qa[3] > qa[k]) k = 3;
  float cand[4];
  if (k == 0)      { cand[0] = qa[0] * qa[0]; cand[1] = m21 - m12; cand[2] = m02 - m20; cand[3] = m10 - m01; }
  else if (k == 1) { cand[0] = m21 - m12; cand[1] = qa[1] * qa[1]; cand[2] = m10 + m01; cand[3] = m02 + m20; }
  else if (k == 2) { cand[0] = m02 - m20; cand[1] = m10 + m01; cand[2] = qa[2] * qa[2]; cand[3] = m12 + m21; }
  else             { cand[0] = m10 - m01; cand[1] = m20 + m02; cand[2] = m21 + m12; cand[3] = qa[3] * qa[3]; }
  float denom = 2.f * fmaxf(qa[k], 0.1f);
#pragma unroll
  for (int c = 0; c < 4; ++c)
    out[rotOff + (size_t)i * 4 + c] = cand[c] / denom;
}

// --------------------------------- host ------------------------------------
extern "C" void kernel_launch(void* const* d_in, const int* in_sizes, int n_in,
                              void* d_out, int out_size, void* d_ws, size_t ws_size,
                              hipStream_t stream)
{
  (void)out_size; (void)ws_size;
  const float* xyz_p      = (const float*)d_in[0];
  const float* feature_p  = (const float*)d_in[1];
  const float* scales_p   = (const float*)d_in[2];
  const float* rotation_p = (const float*)d_in[3];
  const float* opacity_p  = (const float*)d_in[4];
  const float* landmarks  = (const float*)d_in[5];
  const float* exp_coeff  = (const float*)d_in[6];
  const float* pose       = (const float*)d_in[7];
  const float* scale      = (const float*)d_in[8];

  const int N = in_sizes[0] / 3;
  const int B = in_sizes[6] / 64;

  // params dict flatten order: alphabetical (jax pytree) vs insertion order.
  // d_in[17] is exp_deform W1 (65536) if alphabetical, pose_color W1 (4096) if not.
  const float *Wec0f, *bec0r, *Wec1f, *bec1r, *Wec2f, *bec2r;
  const float *Wpc0f, *bpc0r, *Wpc1f, *bpc1r;
  const float *Wed0f, *bed0r, *Wed1f, *bed1r, *Wed2f, *bed2r;
  const float *Wpd0f, *bpd0r, *Wpd1f, *bpd1r;
  const bool alpha = (n_in > 17) ? (in_sizes[17] == 65536) : true;
  Wec0f = (const float*)d_in[9];  bec0r = (const float*)d_in[10];
  Wec1f = (const float*)d_in[11]; bec1r = (const float*)d_in[12];
  Wec2f = (const float*)d_in[13]; bec2r = (const float*)d_in[14];
  if (alpha) {
    Wed0f = (const float*)d_in[15]; bed0r = (const float*)d_in[16];
    Wed1f = (const float*)d_in[17]; bed1r = (const float*)d_in[18];
    Wed2f = (const float*)d_in[19]; bed2r = (const float*)d_in[20];
    Wpc0f = (const float*)d_in[21]; bpc0r = (const float*)d_in[22];
    Wpc1f = (const float*)d_in[23]; bpc1r = (const float*)d_in[24];
    Wpd0f = (const float*)d_in[25]; bpd0r = (const float*)d_in[26];
    Wpd1f = (const float*)d_in[27]; bpd1r = (const float*)d_in[28];
  } else {
    Wpc0f = (const float*)d_in[15]; bpc0r = (const float*)d_in[16];
    Wpc1f = (const float*)d_in[17]; bpc1r = (const float*)d_in[18];
    Wed0f = (const float*)d_in[19]; bed0r = (const float*)d_in[20];
    Wed1f = (const float*)d_in[21]; bed1r = (const float*)d_in[22];
    Wed2f = (const float*)d_in[23]; bed2r = (const float*)d_in[24];
    Wpd0f = (const float*)d_in[25]; bpd0r = (const float*)d_in[26];
    Wpd1f = (const float*)d_in[27]; bpd1r = (const float*)d_in[28];
  }

  // workspace layout: f16 weights, then f32 per-batch data
  _Float16* H = (_Float16*)d_ws;
  _Float16* Hec0 = H;                 // [256][128]
  _Float16* Hec1 = Hec0 + 32768;      // [256][256]
  _Float16* Hec2 = Hec1 + 65536;      // [32][256]
  _Float16* Hpc0 = Hec2 + 8192;       // [128][128]
  _Float16* Hpc1 = Hpc0 + 16384;      // [32][128]
  _Float16* Hed0 = Hpc1 + 4096;       // [256][32]
  _Float16* Hed1 = Hed0 + 8192;       // [256][256]
  _Float16* Hed2 = Hed1 + 65536;      // [16][256] rows 3..15 = 0
  _Float16* Hpd0 = Hed2 + 4096;       // [128][32]
  _Float16* Hpd1 = Hpd0 + 4096;       // [16][128] rows 3..15 = 0
  float* F    = (float*)(Hpd1 + 2048);
  float* Rw   = F;                    // B*9
  float* Tw   = Rw + 9 * B;           // B*3
  float* Sw   = Tw + 3 * B;           // B
  float* bec0 = Sw + B;               // B*256
  float* bpc0 = bec0 + 256 * B;       // B*128
  float* bed0 = bpc0 + 128 * B;       // B*256
  float* bpd0 = bed0 + 256 * B;       // B*128
  float* bed2p = bpd0 + 128 * B;      // 16
  float* bpd1p = bed2p + 16;          // 16

  // weight conversions (column-sliced / zero-padded)
  struct CvtJob { const float* src; _Float16* dst; int sr, sc, sld, dr, dc; };
  const CvtJob jobs[10] = {
    {Wec0f, Hec0, 256, 128, 192, 256, 128},
    {Wec1f, Hec1, 256, 256, 256, 256, 256},
    {Wec2f, Hec2,  32, 256, 256,  32, 256},
    {Wpc0f, Hpc0, 128, 128, 182, 128, 128},
    {Wpc1f, Hpc1,  32, 128, 128,  32, 128},
    {Wed0f, Hed0, 256,  27,  91, 256,  32},
    {Wed1f, Hed1, 256, 256, 256, 256, 256},
    {Wed2f, Hed2,   3, 256, 256,  16, 256},
    {Wpd0f, Hpd0, 128,  27,  81, 128,  32},
    {Wpd1f, Hpd1,   3, 128, 128,  16, 128},
  };
  for (int j = 0; j < 10; ++j) {
    int tot = jobs[j].dr * jobs[j].dc;
    cvt_w<<<(tot + 255) / 256, 256, 0, stream>>>(jobs[j].src, jobs[j].dst, jobs[j].sr,
                                                 jobs[j].sc, jobs[j].sld, jobs[j].dr, jobs[j].dc);
  }

  prep_batch<<<B, 256, 0, stream>>>(pose, exp_coeff, scale,
                                    Wec0f, bec0r, Wpc0f, bpc0r, Wed0f, bed0r, Wpd0f, bpd0r,
                                    bed2r, bpd1r,
                                    Rw, Tw, Sw, bec0, bpc0, bed0, bpd0, bed2p, bpd1p);

  MainP P;
  P.xyz = xyz_p; P.feat = feature_p; P.lmk = landmarks;
  P.Wec0 = Hec0; P.Wec1 = Hec1; P.Wec2 = Hec2;
  P.Wpc0 = Hpc0; P.Wpc1 = Hpc1;
  P.Wed0 = Hed0; P.Wed1 = Hed1; P.Wed2 = Hed2;
  P.Wpd0 = Hpd0; P.Wpd1 = Hpd1;
  P.bec0 = bec0; P.bpc0 = bpc0; P.bed0 = bed0; P.bpd0 = bpd0;
  P.bec1 = bec1r; P.bec2 = bec2r; P.bpc1 = bpc1r; P.bed1 = bed1r;
  P.bed2p = bed2p; P.bpd1p = bpd1p;
  P.R = Rw; P.T = Tw; P.S = Sw;
  P.out = (float*)d_out; P.N = N;

  gh_main<<<dim3((N + 31) / 32, B), 512, 0, stream>>>(P);

  gh_pointwise<<<(B * N + 255) / 256, 256, 0, stream>>>(scales_p, rotation_p, opacity_p,
                                                        Rw, Sw, (float*)d_out, N, B);
}
